// NCAWeightGenerator_89816356094468
// MI455X (gfx1250) — matrix operand order repaired and checked by reading, not compiled, not run on hardware
//
#include <hip/hip_runtime.h>

typedef __attribute__((ext_vector_type(4)))  float    v4f;
typedef __attribute__((ext_vector_type(8)))  float    v8f;
typedef __attribute__((ext_vector_type(16))) _Float16 v16h;
typedef __attribute__((ext_vector_type(8)))  _Float16 v8h;
#define LDB(p) __builtin_shufflevector(*(const v8h*)(p), *(const v8h*)((p) + 16), 0,1,2,3,4,5,6,7,8,9,10,11,12,13,14,15)

#define NCA_C 16
#define NCA_H 128
#define NCA_N 262144
#define NCA_STEPS 50
#define WAVES_PER_WG 4
#define LDS_STRIDE 132  // 132 % 64 == 4 -> conflict-free-ish LDS tile accesses

// split-weight arena (units: halfs), placed at byte offset 16MB inside d_ws
#define W1H_OFF 0       // [128][64]  (K padded 48 -> 64 with zeros)
#define W1L_OFF 8192
#define W2H_OFF 16384   // [128][128]
#define W2L_OFF 32768
#define W3H_OFF 49152   // [16][128]
#define W3L_OFF 51200
#define STATE_FLOATS (NCA_N * NCA_C)

// D(16x16,f32) = A(16x32,f16) * B(32x16,f16) + C   (codegen-confirmed builtin)
__device__ __forceinline__ v8f wmma16(v16h a, v16h b, v8f c) {
  return __builtin_amdgcn_wmma_f32_16x16x32_f16(false, a, false, b, (short)0, c,
                                                false, false);
}

// Markidis 2-term split: x = hi + lo with f16 hi/lo; write into slot idx.
__device__ __forceinline__ void split_put(v16h& hi, v16h& lo, int idx, float x) {
  _Float16 hv = (_Float16)x;
  hi[idx] = hv;
  lo[idx] = (_Float16)(x - (float)hv);
}

// One-time weight split (weights are step-invariant).
__global__ __launch_bounds__(256)
void nca_prep_kernel(const float* __restrict__ W1,   // (128,48)
                     const float* __restrict__ W2,   // (128,128)
                     const float* __restrict__ W3,   // (16,128)
                     _Float16* __restrict__ wsp) {
  const int i = blockIdx.x * 256 + threadIdx.x;      // 0..16383
  if (i < 8192) {                                    // W1 padded to K=64
    const int n = i >> 6, k = i & 63;
    const float x = (k < 48) ? W1[n * 48 + k] : 0.0f;
    const _Float16 hv = (_Float16)x;
    wsp[W1H_OFF + i] = hv;
    wsp[W1L_OFF + i] = (_Float16)(x - (float)hv);
  }
  {                                                  // W2 (16384 elements)
    const float x = W2[i];
    const _Float16 hv = (_Float16)x;
    wsp[W2H_OFF + i] = hv;
    wsp[W2L_OFF + i] = (_Float16)(x - (float)hv);
  }
  if (i < 2048) {                                    // W3
    const float x = W3[i];
    const _Float16 hv = (_Float16)x;
    wsp[W3H_OFF + i] = hv;
    wsp[W3L_OFF + i] = (_Float16)(x - (float)hv);
  }
}

__global__ __launch_bounds__(WAVES_PER_WG * 32)
void nca_step_kernel(const float* __restrict__ src,
                     float* __restrict__ dst,
                     const _Float16* __restrict__ wsp,
                     const float* __restrict__ b1,
                     const float* __restrict__ b2,
                     const float* __restrict__ mu) {
  __shared__ __align__(16) float lds_h[WAVES_PER_WG][16 * LDS_STRIDE];

  const int lane  = threadIdx.x & 31;
  const int wv    = threadIdx.x >> 5;
  const int rhalf = lane & 15;   // M-row (A) / N-col (B,C,D) in the 16-tile
  const int hh    = lane >> 4;   // half-wave selector in the K dimension

  const int row0 = (blockIdx.x * WAVES_PER_WG + wv) * 16;

  // ---- A fragments of P = [s, s-left, s-right] padded to 16x64, split hi/lo.
  // A layout: lane(m,hh): halfs 0-7 = K(hh*8+0..7), halfs 8-15 = K(16+hh*8+0..7)
  const int cell = row0 + rhalf;
  const float* srow = src + (size_t)cell * NCA_C + hh * 8;
  const float* lrow = src + (size_t)((cell - 1) & (NCA_N - 1)) * NCA_C + hh * 8;
  const float* rrow = src + (size_t)((cell + 1) & (NCA_N - 1)) * NCA_C + hh * 8;
  const v4f s0 = *(const v4f*)(srow);
  const v4f s1 = *(const v4f*)(srow + 4);
  const v4f l0 = *(const v4f*)(lrow);
  const v4f l1 = *(const v4f*)(lrow + 4);
  const v4f r0 = *(const v4f*)(rrow);
  const v4f r1 = *(const v4f*)(rrow + 4);

  v16h a0h, a0l, a1h, a1l;   // K-blocks 0 (cols 0-31) and 1 (cols 32-63)
#pragma unroll
  for (int i = 0; i < 4; ++i) {
    split_put(a0h, a0l, i,      s0[i]);             // cols 0-15: state
    split_put(a0h, a0l, 4 + i,  s1[i]);
    split_put(a0h, a0l, 8 + i,  s0[i] - l0[i]);     // cols 16-31: s - left
    split_put(a0h, a0l, 12 + i, s1[i] - l1[i]);
    split_put(a1h, a1l, i,      s0[i] - r0[i]);     // cols 32-47: s - right
    split_put(a1h, a1l, 4 + i,  s1[i] - r1[i]);
    a1h[8 + i]  = (_Float16)0.0f;                   // cols 48-63: zero pad
    a1h[12 + i] = (_Float16)0.0f;
    a1l[8 + i]  = (_Float16)0.0f;
    a1l[12 + i] = (_Float16)0.0f;
  }

  // ---- GEMM1: h1 = relu(P @ W1^T + b1)
  v8f acc[8];
#pragma unroll
  for (int nt = 0; nt < 8; ++nt) {
    const float bv = b1[nt * 16 + rhalf];
#pragma unroll
    for (int j = 0; j < 8; ++j) acc[nt][j] = bv;
  }
  const _Float16* w1h = wsp + W1H_OFF;
  const _Float16* w1l = wsp + W1L_OFF;
#pragma unroll
  for (int b = 0; b < 2; ++b) {
    const v16h ah = b ? a1h : a0h;
    const v16h al = b ? a1l : a0l;
#pragma unroll
    for (int nt = 0; nt < 8; ++nt) {
      const int nrow = nt * 16 + rhalf;
      // B layout: lane(n,hh): halfs 0-7 = K(b*32 + hh*8 + 0..7), halfs 8-15 = K(b*32 + 16 + hh*8 + 0..7), col n
      const v16h bh = LDB(w1h + nrow * 64 + b * 32 + hh * 8);
      const v16h bl = LDB(w1l + nrow * 64 + b * 32 + hh * 8);
      acc[nt] = wmma16(ah, bh, acc[nt]);
      acc[nt] = wmma16(ah, bl, acc[nt]);
      acc[nt] = wmma16(al, bh, acc[nt]);
    }
  }
  // relu + stage h1 (f32, C-layout -> row-major padded tile)
  {
    float* hb = &lds_h[wv][(hh * 8) * LDS_STRIDE + rhalf];
#pragma unroll
    for (int nt = 0; nt < 8; ++nt)
#pragma unroll
      for (int j = 0; j < 8; ++j)
        hb[j * LDS_STRIDE + nt * 16] = fmaxf(acc[nt][j], 0.0f);
  }

  // ---- GEMM2: h2 = relu(h1 @ W2^T + b2)
#pragma unroll
  for (int nt = 0; nt < 8; ++nt) {
    const float bv = b2[nt * 16 + rhalf];
#pragma unroll
    for (int j = 0; j < 8; ++j) acc[nt][j] = bv;
  }
  const _Float16* w2h = wsp + W2H_OFF;
  const _Float16* w2l = wsp + W2L_OFF;
  const float* hrow = &lds_h[wv][rhalf * LDS_STRIDE + hh * 8];
#pragma unroll
  for (int b = 0; b < 4; ++b) {
    const v4f x0 = *(const v4f*)(hrow + b * 32);
    const v4f x1 = *(const v4f*)(hrow + b * 32 + 4);
    const v4f y0 = *(const v4f*)(hrow + b * 32 + 16);
    const v4f y1 = *(const v4f*)(hrow + b * 32 + 20);
    v16h ah, al;
#pragma unroll
    for (int i = 0; i < 4; ++i) {
      split_put(ah, al, i,      x0[i]);
      split_put(ah, al, 4 + i,  x1[i]);
      split_put(ah, al, 8 + i,  y0[i]);
      split_put(ah, al, 12 + i, y1[i]);
    }
#pragma unroll
    for (int nt = 0; nt < 8; ++nt) {
      const int nrow = nt * 16 + rhalf;
      const v16h bh = LDB(w2h + nrow * 128 + b * 32 + hh * 8);
      const v16h bl = LDB(w2l + nrow * 128 + b * 32 + hh * 8);
      acc[nt] = wmma16(ah, bh, acc[nt]);
      acc[nt] = wmma16(ah, bl, acc[nt]);
      acc[nt] = wmma16(al, bh, acc[nt]);
    }
  }
  // relu + re-stage h2 (same-wave DS ops are in-order; safe to overwrite h1)
  {
    float* hb = &lds_h[wv][(hh * 8) * LDS_STRIDE + rhalf];
#pragma unroll
    for (int nt = 0; nt < 8; ++nt)
#pragma unroll
      for (int j = 0; j < 8; ++j)
        hb[j * LDS_STRIDE + nt * 16] = fmaxf(acc[nt][j], 0.0f);
  }

  // ---- GEMM3: delta = h2 @ W3^T  (16x16)
  const _Float16* w3h = wsp + W3H_OFF;
  const _Float16* w3l = wsp + W3L_OFF;
  v8f d;
#pragma unroll
  for (int j = 0; j < 8; ++j) d[j] = 0.0f;
#pragma unroll
  for (int b = 0; b < 4; ++b) {
    const v4f x0 = *(const v4f*)(hrow + b * 32);
    const v4f x1 = *(const v4f*)(hrow + b * 32 + 4);
    const v4f y0 = *(const v4f*)(hrow + b * 32 + 16);
    const v4f y1 = *(const v4f*)(hrow + b * 32 + 20);
    v16h ah, al;
#pragma unroll
    for (int i = 0; i < 4; ++i) {
      split_put(ah, al, i,      x0[i]);
      split_put(ah, al, 4 + i,  x1[i]);
      split_put(ah, al, 8 + i,  y0[i]);
      split_put(ah, al, 12 + i, y1[i]);
    }
    const v16h bh = LDB(w3h + rhalf * 128 + b * 32 + hh * 8);
    const v16h bl = LDB(w3l + rhalf * 128 + b * 32 + hh * 8);
    d = wmma16(ah, bh, d);
    d = wmma16(ah, bl, d);
    d = wmma16(al, bh, d);
  }

  // ---- epilogue: dst = src + delta * (mu < 0.5)
  const int mrow0 = row0 + hh * 8;
#pragma unroll
  for (int j = 0; j < 8; ++j) {
    const int row = mrow0 + j;
    const float m = (mu[row] < 0.5f) ? 1.0f : 0.0f;
    const size_t idx = (size_t)row * NCA_C + rhalf;
    dst[idx] = src[idx] + d[j] * m;
  }
}

extern "C" void kernel_launch(void* const* d_in, const int* in_sizes, int n_in,
                              void* d_out, int out_size, void* d_ws, size_t ws_size,
                              hipStream_t stream) {
  (void)in_sizes; (void)n_in; (void)out_size; (void)ws_size;
  const float* state0 = (const float*)d_in[0];
  const float* W1 = (const float*)d_in[1];
  const float* b1 = (const float*)d_in[2];
  const float* W2 = (const float*)d_in[3];
  const float* b2 = (const float*)d_in[4];
  const float* W3 = (const float*)d_in[5];
  const float* mu = (const float*)d_in[6];
  float* out = (float*)d_out;
  float* ws  = (float*)d_ws;                               // 16MB state ping buffer
  _Float16* wsp = (_Float16*)((char*)d_ws + (size_t)STATE_FLOATS * 4);

  // One-time (per launch) weight split; deterministic, cheap (16K threads).
  nca_prep_kernel<<<64, 256, 0, stream>>>(W1, W2, W3, wsp);

  const int blocks  = NCA_N / (WAVES_PER_WG * 16);         // 4096
  const int threads = WAVES_PER_WG * 32;                   // 128

  // Ping-pong ws <-> out; 50 steps (even) lands the final state in d_out.
  for (int s = 0; s < NCA_STEPS; ++s) {
    const float* src = (s == 0) ? state0 : ((s & 1) ? ws : out);
    float* dst = (s & 1) ? out : ws;
    nca_step_kernel<<<blocks, threads, 0, stream>>>(
        src, dst, wsp, b1, b2, mu + (size_t)s * NCA_N);
  }
}
